// Decoder_3753801417244
// MI455X (gfx1250) — matrix operand order repaired and checked by reading, not compiled, not run on hardware
//
#include <hip/hip_runtime.h>

#define V_ 32000
#define E_ 1024
#define H_ 2048
#define B_ 512

typedef __attribute__((ext_vector_type(16))) __bf16 v16bf;
typedef __attribute__((ext_vector_type(8)))  float  v8f;

struct __align__(16) U4 { unsigned int x, y, z, w; };
struct U8x { U4 lo, hi; };
struct __align__(8) U2 { unsigned int a, b; };

// Pack two fp32 -> two bf16 (round-to-nearest) in 3 VALU:
// two v_add_u32 (+0x8000) then one v_perm_b32 picking the high halves.
__device__ __forceinline__ unsigned int pkbf(float lo, float hi) {
    unsigned int a = __builtin_bit_cast(unsigned int, lo) + 0x8000u;
    unsigned int b = __builtin_bit_cast(unsigned int, hi) + 0x8000u;
    // result bytes: {b[3],b[2],a[3],a[2]} -> (bf(hi)<<16)|bf(lo)
    return __builtin_amdgcn_perm(b, a, 0x07060302u);
}

// store 4 consecutive bf16 as one 8-byte LDS store
__device__ __forceinline__ void store_bf4(unsigned short* dst, float4 v) {
    U2 p;
    p.a = pkbf(v.x, v.y);
    p.b = pkbf(v.z, v.w);
    *(U2*)dst = p;
}

// A-matrix 16x32 bf16 fragment (ISA 7.12.2): lane(0-15)=row M, hi=lane>>4
// elements [0..7] = K hi*8..hi*8+7 ; [8..15] = K 16+hi*8..16+hi*8+7
__device__ __forceinline__ v16bf frag_a(const unsigned short* row, int hi) {
    U8x t;
    t.lo = *(const U4*)(row + hi * 8);
    t.hi = *(const U4*)(row + 16 + hi * 8);
    return __builtin_bit_cast(v16bf, t);
}

// B-matrix 32x16 bf16 fragment: lane = khalf*16 + n; [0..7] = K khalf*8.., [8..15] = K 16+khalf*8..
__device__ __forceinline__ v16bf frag_b(const unsigned short* row, int khalf) {
    U8x t;
    t.lo = *(const U4*)(row + khalf * 8);
    t.hi = *(const U4*)(row + 16 + khalf * 8);
    return __builtin_bit_cast(v16bf, t);
}

__device__ __forceinline__ v8f wmma_bf16(v16bf a, v16bf b, v8f c) {
    return __builtin_amdgcn_wmma_f32_16x16x32_bf16(
        /*neg_a=*/false, a, /*neg_b=*/false, b,
        /*c_mod=*/(short)0, c, /*reuse_a=*/false, /*reuse_b=*/false);
}

__device__ __forceinline__ float sigmoidf_(float x) {
    return 1.0f / (1.0f + __expf(-x));
}
__device__ __forceinline__ float tanhf_(float x) {
    float e2 = __expf(2.0f * x);
    return (e2 - 1.0f) / (e2 + 1.0f);
}

// ---------------------------------------------------------------------------
// Kernel 1: fused embedding gather + GRU cell -> h_new [B, H]
// 256 thr (8 waves), block tile M=32 x N=64; wave = one 16x16 h_new tile.
// Single 96-step pipelined K loop: steps 0..31 = W_ih over E (A = emb gather),
// steps 32..95 = W_hh over H (A = hidden). Double-buffered LDS, 1 barrier/step.
// Grid: x = M (fastest) so co-resident blocks share the same weight tile in L2.
// ---------------------------------------------------------------------------
__global__ __launch_bounds__(256) void gru_step_kernel(
    const int*   __restrict__ input_step,
    const float* __restrict__ hidden,
    const float* __restrict__ emb,
    const float* __restrict__ W_ih,
    const float* __restrict__ W_hh,
    const float* __restrict__ b_ih,
    const float* __restrict__ b_hh,
    float*       __restrict__ h_new)
{
    __shared__ __align__(16) unsigned short As[2][32][40];
    __shared__ __align__(16) unsigned short Ws[2][3][64][40];

    const int tid  = threadIdx.x;
    const int lane = tid & 31;
    const int wave = tid >> 5;
    const int lm   = lane & 15;
    const int lhi  = lane >> 4;

    const int bBlk = blockIdx.x * 32;   // batch base   (fastest-varying)
    const int nBlk = blockIdx.y * 64;   // hidden-col base
    const int m0   = (wave >> 2) * 16;
    const int n0   = (wave & 3) * 16;

    v8f accR = {}, accZ = {}, accGI = {}, accGH = {};

    // Cooperative staging indices (all 32-bit offsets -> saddr+voffset mode)
    const int am = tid >> 3;            // A row 0..31
    const int ak = (tid & 7) * 4;       // A col quad

    const unsigned tok   = (unsigned)input_step[bBlk + am];
    const unsigned aOffE = tok * (unsigned)E_ + (unsigned)ak;
    const unsigned aOffH = (unsigned)(bBlk + am) * (unsigned)H_ + (unsigned)ak;

    int gq[6], nq[6], kqv[6];
    unsigned wOffE[6], wOffH[6];
    #pragma unroll
    for (int q = 0; q < 6; ++q) {
        int idx = tid + q * 256;        // 3 gates x 64 rows x 8 quads = 1536
        gq[q]  = idx >> 9;
        nq[q]  = (idx & 511) >> 3;
        kqv[q] = (idx & 7) * 4;
        unsigned row = (unsigned)(gq[q] * H_ + nBlk + nq[q]);
        wOffE[q] = row * (unsigned)E_ + (unsigned)kqv[q];
        wOffH[q] = row * (unsigned)H_ + (unsigned)kqv[q];
    }

    float4 sa;          // staged A quad
    float4 sw[6];       // staged W quads

    auto load_stage = [&](int it) {
        const bool ph = it >= 32;
        const unsigned k0 = (unsigned)(ph ? (it - 32) : it) * 32u;
        sa = *(const float4*)((ph ? hidden : emb) + ((ph ? aOffH : aOffE) + k0));
        const float* Wsrc = ph ? W_hh : W_ih;
        #pragma unroll
        for (int q = 0; q < 6; ++q)
            sw[q] = *(const float4*)(Wsrc + ((ph ? wOffH[q] : wOffE[q]) + k0));
    };
    auto store_stage = [&](int bsel) {
        store_bf4(&As[bsel][am][ak], sa);
        #pragma unroll
        for (int q = 0; q < 6; ++q)
            store_bf4(&Ws[bsel][gq[q]][nq[q]][kqv[q]], sw[q]);
    };

    load_stage(0);
    store_stage(0);
    int buf = 0;
    for (int it = 0; it < 96; ++it) {
        __syncthreads();                        // buf ready for all waves
        const bool more = (it + 1) < 96;
        if (more) load_stage(it + 1);           // global loads overlap compute

        v16bf af = frag_a(&As[buf][m0 + lm][0], lhi);
        v16bf b0 = frag_b(&Ws[buf][0][n0 + lm][0], lhi);
        v16bf b1 = frag_b(&Ws[buf][1][n0 + lm][0], lhi);
        v16bf b2 = frag_b(&Ws[buf][2][n0 + lm][0], lhi);
        accR = wmma_bf16(af, b0, accR);         // gi_r + gh_r fused
        accZ = wmma_bf16(af, b1, accZ);         // gi_z + gh_z fused
        if (it < 32) accGI = wmma_bf16(af, b2, accGI);
        else         accGH = wmma_bf16(af, b2, accGH);

        if (more) store_stage(buf ^ 1);         // fill other buffer
        buf ^= 1;
    }

    // Epilogue: C/D layout — VGPR i, lane: M = i + 8*lhi, N = lm
    const int j = nBlk + n0 + lm;
    const float br  = b_ih[j]        + b_hh[j];
    const float bz  = b_ih[j + H_]   + b_hh[j + H_];
    const float bin = b_ih[j + 2*H_];
    const float bhn = b_hh[j + 2*H_];
    #pragma unroll
    for (int i = 0; i < 8; ++i) {
        const int b = bBlk + m0 + i + lhi * 8;
        const unsigned o = (unsigned)b * (unsigned)H_ + (unsigned)j;
        float r  = sigmoidf_(accR[i] + br);
        float z  = sigmoidf_(accZ[i] + bz);
        float n  = tanhf_((accGI[i] + bin) + r * (accGH[i] + bhn));
        h_new[o] = (1.0f - z) * n + z * hidden[o];   // regular TH: reused by k2
    }
}

// ---------------------------------------------------------------------------
// Kernel 2: logits = h_new @ fc_W.T + fc_b   [B, V]
// 256 thr (8 waves), block tile M=32 x N=128; wave owns two 16x16 tiles.
// Double-buffered LDS, 1 barrier per K-step, prefetch of the +2 W tile.
// Grid: x = M (fastest) -> the 16 blocks sharing one fc_W tile are
// co-resident: fc_W streams from HBM exactly once (~262 MB, ~11 us floor).
// Logits stores are non-temporal (write-once) so they don't evict fc_W in L2.
// ---------------------------------------------------------------------------
__global__ __launch_bounds__(256) void logits_kernel(
    const float* __restrict__ h_new,
    const float* __restrict__ fc_W,
    const float* __restrict__ fc_b,
    float*       __restrict__ logits)
{
    __shared__ __align__(16) unsigned short As[2][32][40];
    __shared__ __align__(16) unsigned short Ws[2][128][40];

    const int tid  = threadIdx.x;
    const int lane = tid & 31;
    const int wave = tid >> 5;
    const int lm   = lane & 15;
    const int lhi  = lane >> 4;

    const int bBlk = blockIdx.x * 32;    // batch base (fastest-varying)
    const int vBlk = blockIdx.y * 128;   // vocab base
    const int m0   = (wave >> 2) * 16;
    const int n0   = (wave & 3) * 32;

    v8f acc0 = {}, acc1 = {};

    const int am = tid >> 3;
    const int ak = (tid & 7) * 4;
    const unsigned aOff = (unsigned)(bBlk + am) * (unsigned)H_ + (unsigned)ak;

    int nq[4], kqv[4];
    unsigned wOff[4];
    #pragma unroll
    for (int q = 0; q < 4; ++q) {
        int idx = tid + q * 256;        // 128 rows x 8 quads = 1024
        nq[q]  = idx >> 3;
        kqv[q] = (idx & 7) * 4;
        wOff[q] = (unsigned)(vBlk + nq[q]) * (unsigned)H_ + (unsigned)kqv[q];
    }

    float4 sa;
    float4 sw[4];

    auto load_stage = [&](int it) {
        const unsigned k0 = (unsigned)it * 32u;
        sa = *(const float4*)(h_new + (aOff + k0));
        #pragma unroll
        for (int q = 0; q < 4; ++q)
            sw[q] = *(const float4*)(fc_W + (wOff[q] + k0));
        // pull the +2 step's tile toward the cache (global_prefetch_b8)
        if (k0 + 64u < (unsigned)H_)
            __builtin_prefetch(fc_W + (wOff[0] + k0 + 64u), 0, 0);
    };
    auto store_stage = [&](int bsel) {
        store_bf4(&As[bsel][am][ak], sa);
        #pragma unroll
        for (int q = 0; q < 4; ++q)
            store_bf4(&Ws[bsel][nq[q]][kqv[q]], sw[q]);
    };

    load_stage(0);
    store_stage(0);
    int buf = 0;
    for (int it = 0; it < H_ / 32; ++it) {
        __syncthreads();
        const bool more = (it + 1) < (H_ / 32);
        if (more) load_stage(it + 1);

        v16bf af  = frag_a(&As[buf][m0 + lm][0], lhi);
        v16bf bf0 = frag_b(&Ws[buf][n0 + lm][0], lhi);
        v16bf bf1 = frag_b(&Ws[buf][n0 + 16 + lm][0], lhi);
        acc0 = wmma_bf16(af, bf0, acc0);
        acc1 = wmma_bf16(af, bf1, acc1);

        if (more) store_stage(buf ^ 1);
        buf ^= 1;
    }

    const int v0 = vBlk + n0 + lm;
    const int v1 = v0 + 16;
    const float fb0 = fc_b[v0];
    const float fb1 = fc_b[v1];
    #pragma unroll
    for (int i = 0; i < 8; ++i) {
        const unsigned b = (unsigned)(bBlk + m0 + i + lhi * 8);
        __builtin_nontemporal_store(acc0[i] + fb0,
                                    &logits[b * (unsigned)V_ + (unsigned)v0]);
        __builtin_nontemporal_store(acc1[i] + fb1,
                                    &logits[b * (unsigned)V_ + (unsigned)v1]);
    }
}

extern "C" void kernel_launch(void* const* d_in, const int* in_sizes, int n_in,
                              void* d_out, int out_size, void* d_ws, size_t ws_size,
                              hipStream_t stream) {
    (void)in_sizes; (void)n_in; (void)out_size; (void)d_ws; (void)ws_size;

    const int*   input_step = (const int*)  d_in[0];
    const float* hidden     = (const float*)d_in[1];
    const float* emb        = (const float*)d_in[2];
    const float* W_ih       = (const float*)d_in[3];
    const float* W_hh       = (const float*)d_in[4];
    const float* b_ih       = (const float*)d_in[5];
    const float* b_hh       = (const float*)d_in[6];
    const float* fc_W       = (const float*)d_in[7];
    const float* fc_b       = (const float*)d_in[8];

    float* logits = (float*)d_out;                    // [B, V] first
    float* h_new  = logits + (size_t)B_ * V_;         // then hidden [1, B, H]

    // M (batch) on blockIdx.x so weight tiles are shared by co-resident blocks
    dim3 g1(B_ / 32, H_ / 64);       // (16, 32)
    gru_step_kernel<<<g1, dim3(256), 0, stream>>>(
        input_step, hidden, emb, W_ih, W_hh, b_ih, b_hh, h_new);

    dim3 g2(B_ / 32, V_ / 128);      // (16, 250)
    logits_kernel<<<g2, dim3(256), 0, stream>>>(h_new, fc_W, fc_b, logits);
}
